// CausalTemporalAttention_89163521065836
// MI455X (gfx1250) — matrix operand order repaired and checked by reading, not compiled, not run on hardware
//
#include <hip/hip_runtime.h>
#include <hip/hip_bf16.h>

// ---------------------------------------------------------------------------
// CausalTemporalAttention on gfx1250 (MI455X), wave32 + WMMA f16->f32.
//   DIM=1024, HEADS=16, HEAD_DIM=64, B=2, T=2048
// Pipeline:
//   1) QKV GEMM  (x[4096x1024] @ w_qkv[1024x3072]) -> Qh/Kh (B,H,T,64) f16,
//      Vt (B,H,64,T) f16.  pos_bias + 1/sqrt(64) folded into Q.
//   2) Flash attention (one wave per 16 query rows), WMMA for S, PV and the
//      softmax row-sum (P x ones); lane butterflies only for the row max.
//   3) Out GEMM  (O[4096x1024] @ w_out[1024x1024]) + b_out -> d_out f32.
// ---------------------------------------------------------------------------

typedef _Float16 v8h  __attribute__((ext_vector_type(8)));
typedef _Float16 v16h __attribute__((ext_vector_type(16)));
typedef float    v8f  __attribute__((ext_vector_type(8)));
typedef float    v4f  __attribute__((ext_vector_type(4)));

#define CDIM       1024
#define CT         2048
#define CH         16
#define CDH        64
#define CB         2
#define CBT        (CB*CT)         // 4096
#define LDS_STRIDE 40              // halves per row (32 data + 8 pad), 80B: 16B aligned

// D = A*B + C, f16 inputs, f32 accum (V_WMMA_F32_16X16X32_F16)
__device__ __forceinline__ v8f wmma_f16(v16h a, v16h b, v8f c) {
  return __builtin_amdgcn_wmma_f32_16x16x32_f16(false, a, false, b, (short)0, c,
                                                false, false);
}

// Assemble a 16-half fragment from two 16B chunks at p and p+off (in halves).
__device__ __forceinline__ v16h ld_frag(const _Float16* p, int off) {
  v8h lo = *(const v8h*)p;
  v8h hi = *(const v8h*)(p + off);
  v16h r;
#pragma unroll
  for (int i = 0; i < 8; ++i) { r[i] = lo[i]; r[i + 8] = hi[i]; }
  return r;
}

__device__ __forceinline__ float rmax16(float v) {
#pragma unroll
  for (int m = 1; m < 16; m <<= 1) v = fmaxf(v, __shfl_xor(v, m, 32));
  return v;
}

// ---------------------------------------------------------------------------
// Generic 128x128-tile GEMM:  C[M,N] = A[M,K] * B[K,N]  (A,B f32 in memory,
// converted to f16 in LDS; f32 WMMA accumulation).  MODE 0: QKV epilogue,
// MODE 1: out-projection epilogue.
// ---------------------------------------------------------------------------
template <int MODE>
__global__ __launch_bounds__(256) void gemm_ws_kernel(
    const float* __restrict__ A, const float* __restrict__ B,
    const float* __restrict__ bias, const float* __restrict__ pos_bias,
    _Float16* __restrict__ Qh, _Float16* __restrict__ Kh,
    _Float16* __restrict__ Vt, float* __restrict__ outF, int ldb) {
  __shared__ _Float16 lsA[128 * LDS_STRIDE];
  __shared__ _Float16 lsB[128 * LDS_STRIDE];

  const int tid  = threadIdx.x;
  const int lane = tid & 31;
  const int wave = tid >> 5;
  const int wm   = wave & 1;   // 0..1  (64 rows each)
  const int wn   = wave >> 1;  // 0..3  (32 cols each)
  const int lm   = lane & 15;
  const int hiH  = (lane >> 4);              // 0 or 1
  const int n0   = blockIdx.x * 128;
  const int m0   = blockIdx.y * 128;

  v8f acc[4][2];
#pragma unroll
  for (int mt = 0; mt < 4; ++mt)
#pragma unroll
    for (int nt = 0; nt < 2; ++nt) acc[mt][nt] = (v8f)0.0f;

  for (int k0 = 0; k0 < CDIM; k0 += 32) {
    // --- stage A tile (128x32) row-major, f32 -> f16 ---
#pragma unroll
    for (int it = 0; it < 4; ++it) {
      int e   = it * 256 + tid;        // 0..1023 float4s
      int row = e >> 3;                // 8 float4 per row
      int kq  = (e & 7) << 2;
      v4f v = *(const v4f*)(A + (size_t)(m0 + row) * CDIM + k0 + kq);
      _Float16* d = &lsA[row * LDS_STRIDE + kq];
      d[0] = (_Float16)v[0]; d[1] = (_Float16)v[1];
      d[2] = (_Float16)v[2]; d[3] = (_Float16)v[3];
    }
    // --- stage B tile (32x128) transposed -> lsB[n][k], f32 -> f16 ---
#pragma unroll
    for (int it = 0; it < 16; ++it) {
      int e = it * 256 + tid;          // 0..4095
      int n = e & 127;                 // consecutive tid -> consecutive n
      int k = e >> 7;
      float v = B[(size_t)(k0 + k) * ldb + n0 + n];
      lsB[n * LDS_STRIDE + k] = (_Float16)v;
    }
    __syncthreads();

    v16h af[4], bf[2];
#pragma unroll
    for (int mt = 0; mt < 4; ++mt) {
      const _Float16* pa =
          &lsA[(wm * 64 + mt * 16 + lm) * LDS_STRIDE + (hiH ? 8 : 0)];
      af[mt] = ld_frag(pa, 16);
    }
#pragma unroll
    for (int nt = 0; nt < 2; ++nt) {
      const _Float16* pb =
          &lsB[(wn * 32 + nt * 16 + lm) * LDS_STRIDE + (hiH ? 8 : 0)];
      bf[nt] = ld_frag(pb, 16);
    }
#pragma unroll
    for (int mt = 0; mt < 4; ++mt)
#pragma unroll
      for (int nt = 0; nt < 2; ++nt)
        acc[mt][nt] = wmma_f16(af[mt], bf[nt], acc[mt][nt]);
    __syncthreads();
  }

  // ---- epilogue ----
#pragma unroll
  for (int mt = 0; mt < 4; ++mt) {
#pragma unroll
    for (int nt = 0; nt < 2; ++nt) {
#pragma unroll
      for (int r = 0; r < 8; ++r) {
        int m = m0 + wm * 64 + mt * 16 + r + (hiH << 3);
        int n = n0 + wn * 32 + nt * 16 + lm;
        float val = acc[mt][nt][r] + bias[n];
        if (MODE == 0) {
          // n in [0,3072): which matrix / head / dim
          int which = n >> 10;           // 0=q 1=k 2=v
          int c     = n & 1023;
          int h     = c >> 6;
          int d     = c & 63;
          int b     = m >> 11;           // T = 2048
          int t     = m & 2047;
          size_t bh = (size_t)(b * CH + h);
          if (which == 0) {
            val = (val + pos_bias[c]) * 0.125f;  // fold scale = HEAD_DIM^-0.5
            Qh[(bh * CT + t) * CDH + d] = (_Float16)val;
          } else if (which == 1) {
            Kh[(bh * CT + t) * CDH + d] = (_Float16)val;
          } else {
            Vt[(bh * CDH + d) * CT + t] = (_Float16)val;  // transposed
          }
        } else {
          outF[(size_t)m * CDIM + n] = val;
        }
      }
    }
  }
}

// ---------------------------------------------------------------------------
// Flash attention: grid = B*H*(T/128), block = 256 (8 waves).
// Each wave owns 16 query rows; streams 32-key chunks with online softmax.
// Row-sum of P is computed by WMMA against an all-ones B fragment.
// ---------------------------------------------------------------------------
__global__ __launch_bounds__(256) void attn_kernel(
    const _Float16* __restrict__ Qh, const _Float16* __restrict__ Kh,
    const _Float16* __restrict__ Vt, float* __restrict__ Of) {
  __shared__ _Float16 lsP[8 * 16 * LDS_STRIDE];  // per-wave 16x32 P tile

  const int tid  = threadIdx.x;
  const int lane = tid & 31;
  const int wave = tid >> 5;
  const int lm   = lane & 15;
  const int hiH  = lane >> 4;
  const int bh   = blockIdx.x >> 4;          // b*16+h
  const int q0   = (blockIdx.x & 15) * 128 + wave * 16;
  _Float16* ldsP = &lsP[wave * 16 * LDS_STRIDE];

  const _Float16* Qb = Qh + (size_t)bh * CT * CDH;
  const _Float16* Kb = Kh + (size_t)bh * CT * CDH;
  const _Float16* Vb = Vt + (size_t)bh * CDH * CT;

  // Q fragments (A-layout), one per 32-wide d chunk
  v16h qf[2];
#pragma unroll
  for (int c = 0; c < 2; ++c) {
    const _Float16* p =
        Qb + (size_t)(q0 + lm) * CDH + c * 32 + (hiH ? 8 : 0);
    qf[c] = ld_frag(p, 16);
  }

  v16h onesf;            // all-ones B fragment: D = P x ones -> row sums
#pragma unroll
  for (int i = 0; i < 16; ++i) onesf[i] = (_Float16)1.0f;

  v8f o[4];
#pragma unroll
  for (int t = 0; t < 4; ++t) o[t] = (v8f)0.0f;
  float rmax[8], rsum[8];
#pragma unroll
  for (int r = 0; r < 8; ++r) { rmax[r] = -1e30f; rsum[r] = 0.0f; }

  for (int j = 0; j <= q0 + 15; j += 32) {
    // ---- S = Q K^T for keys [j, j+32) ----
    v8f s0 = (v8f)0.0f, s1 = (v8f)0.0f;
#pragma unroll
    for (int c = 0; c < 2; ++c) {
      const _Float16* pk0 =
          Kb + (size_t)(j + lm) * CDH + c * 32 + (hiH ? 8 : 0);
      const _Float16* pk1 =
          Kb + (size_t)(j + 16 + lm) * CDH + c * 32 + (hiH ? 8 : 0);
      s0 = wmma_f16(qf[c], ld_frag(pk0, 16), s0);
      s1 = wmma_f16(qf[c], ld_frag(pk1, 16), s1);
    }
    // ---- causal mask (wave-uniform guard: only diagonal chunks pay) ----
    if (j + 31 > q0) {
#pragma unroll
      for (int r = 0; r < 8; ++r) {
        int qi = q0 + r + (hiH << 3);
        if (j + lm > qi)      s0[r] = -1e30f;
        if (j + 16 + lm > qi) s1[r] = -1e30f;
      }
    }
    // ---- online softmax: row max by butterfly, exp, stage P to LDS ----
    float corr[8];
#pragma unroll
    for (int r = 0; r < 8; ++r) {
      int m  = r + (hiH << 3);
      float cm = rmax16(fmaxf(s0[r], s1[r]));
      float nm = fmaxf(rmax[r], cm);
      float cf = __expf(rmax[r] - nm);
      float p0 = __expf(s0[r] - nm);
      float p1 = __expf(s1[r] - nm);
      rmax[r] = nm;
      corr[r] = cf;
      ldsP[m * LDS_STRIDE + lm]      = (_Float16)p0;
      ldsP[m * LDS_STRIDE + 16 + lm] = (_Float16)p1;
    }
    // ---- P fragment (A-layout) from per-wave LDS slice ----
    v16h pf = ld_frag(&ldsP[lm * LDS_STRIDE + (hiH ? 8 : 0)], 16);
    // ---- row sums via WMMA: every lane gets its own row's sum ----
    v8f cs = wmma_f16(pf, onesf, (v8f)0.0f);
#pragma unroll
    for (int r = 0; r < 8; ++r) rsum[r] = rsum[r] * corr[r] + cs[r];
#pragma unroll
    for (int t = 0; t < 4; ++t)
#pragma unroll
      for (int r = 0; r < 8; ++r) o[t][r] *= corr[r];

    // ---- O += P V ----
#pragma unroll
    for (int t = 0; t < 4; ++t) {
      const _Float16* pv =
          Vb + (size_t)(t * 16 + lm) * CT + j + (hiH ? 8 : 0);
      o[t] = wmma_f16(pf, ld_frag(pv, 16), o[t]);
    }
  }

  // ---- normalize + write back as f32 [B,T,DIM] with c = h*64+d ----
  const int b = bh >> 4, h = bh & 15;
#pragma unroll
  for (int t = 0; t < 4; ++t) {
#pragma unroll
    for (int r = 0; r < 8; ++r) {
      int m = r + (hiH << 3);
      int d = t * 16 + lm;
      Of[((size_t)(b * CT) + q0 + m) * CDIM + h * CDH + d] = o[t][r] / rsum[r];
    }
  }
}

extern "C" void kernel_launch(void* const* d_in, const int* in_sizes, int n_in,
                              void* d_out, int out_size, void* d_ws, size_t ws_size,
                              hipStream_t stream) {
  const float* x        = (const float*)d_in[0];
  const float* w_qkv    = (const float*)d_in[1];
  const float* b_qkv    = (const float*)d_in[2];
  const float* w_out    = (const float*)d_in[3];
  const float* b_out    = (const float*)d_in[4];
  const float* pos_bias = (const float*)d_in[5];

  char* ws = (char*)d_ws;
  _Float16* Qh = (_Float16*)ws;                       // 4M halves = 8 MB
  _Float16* Kh = Qh + (size_t)CB * CH * CT * CDH;     // 8 MB
  _Float16* Vt = Kh + (size_t)CB * CH * CT * CDH;     // 8 MB
  float*    Of = (float*)(ws + 3ull * 8388608);       // 16 MB

  // 1) QKV projection: M=4096, N=3072, K=1024
  dim3 g1(3072 / 128, CBT / 128);
  gemm_ws_kernel<0><<<g1, 256, 0, stream>>>(x, w_qkv, b_qkv, pos_bias,
                                            Qh, Kh, Vt, nullptr, 3072);

  // 2) Flash attention: B*H * (T/128) workgroups
  attn_kernel<<<CB * CH * (CT / 128), 256, 0, stream>>>(Qh, Kh, Vt, Of);

  // 3) Output projection: M=4096, N=1024, K=1024
  dim3 g3(CDIM / 128, CBT / 128);
  gemm_ws_kernel<1><<<g3, 256, 0, stream>>>(Of, w_out, b_out, nullptr,
                                            nullptr, nullptr, nullptr,
                                            (float*)d_out, CDIM);
}
